// CutsEmbedding_77309411328401
// MI455X (gfx1250) — matrix operand order repaired and checked by reading, not compiled, not run on hardware
//
#include <hip/hip_runtime.h>
#include <hip/hip_bf16.h>

// ---------------- WMMA helper types ----------------
typedef __attribute__((ext_vector_type(16))) _Float16 v16h;
typedef __attribute__((ext_vector_type(8)))  _Float16 v8h;
typedef __attribute__((ext_vector_type(4)))  _Float16 v4h;
typedef __attribute__((ext_vector_type(8)))  float    v8f;

__device__ __forceinline__ v8f wmma_f16(v16h a, v16h b, v8f c) {
  // D = A(16x32 f16) * B(32x16 f16) + C(16x16 f32)
  return __builtin_amdgcn_wmma_f32_16x16x32_f16(false, a, false, b, (short)0, c, false, false);
}

// A operand: LDS row-major [16][stride] f16. Lane L holds row L&15,
// K = kOff + {8h..8h+7, 8h+16..8h+23} where h = L>>4.  (ISA 16-bit A layout)
__device__ __forceinline__ v16h ldsA(const _Float16* base, int stride, int kOff, int lane) {
  const _Float16* p = base + (lane & 15) * stride + kOff + ((lane >> 4) << 3);
  union { v16h v; v8h h[2]; } u;
  u.h[0] = *(const v8h*)(p);
  u.h[1] = *(const v8h*)(p + 16);
  return u.v;
}

// B operand: weights transposed in LDS as Wt[n][k] (k contiguous, stride halves).
// Lane L holds column n = (L&15)+16*nhalf, K = kOff + {8h..8h+7, 8h+16..8h+23} where h = L>>4.
__device__ __forceinline__ v16h ldsB(const _Float16* wt, int stride, int kOff, int nhalf, int lane) {
  const _Float16* p = wt + ((lane & 15) + (nhalf << 4)) * stride + kOff + ((lane >> 4) << 3);
  union { v16h v; v8h h[2]; } u;
  u.h[0] = *(const v8h*)(p);
  u.h[1] = *(const v8h*)(p + 16);
  return u.v;
}

// Store fp32 accumulator (16x16 D tile, nhalf selects cols 0-15/16-31)
// to LDS f16 row-major [16][stride], optional ReLU.
__device__ __forceinline__ void storeD_f16(v8f d, _Float16* out, int stride, int nhalf,
                                           int lane, bool relu) {
  int n  = (lane & 15) + (nhalf << 4);
  int mb = (lane >> 4) << 3;
#pragma unroll
  for (int r = 0; r < 8; ++r) {
    float x = d[r];
    if (relu) x = fmaxf(x, 0.0f);
    out[(mb + r) * stride + n] = (_Float16)x;
  }
}

// 2-layer MLP on a 16-row tile: A[16][KPAD](LDS f16) -> ReLU hidden[16][32] -> out 16x32 (o0,o1)
template <int KPAD>
__device__ __forceinline__ void mlp2(const _Float16* A, const _Float16* W1t, const float* b1,
                                     const _Float16* W2t, const float* b2, _Float16* hid,
                                     int lane, v8f& o0, v8f& o1) {
#pragma unroll
  for (int nh = 0; nh < 2; ++nh) {
    float cb = b1[(lane & 15) + (nh << 4)];
    v8f c = {cb, cb, cb, cb, cb, cb, cb, cb};
#pragma unroll
    for (int kc = 0; kc < KPAD; kc += 32)
      c = wmma_f16(ldsA(A, KPAD, kc, lane), ldsB(W1t, KPAD, kc, nh, lane), c);
    storeD_f16(c, hid, 32, nh, lane, true);
  }
  v16h a = ldsA(hid, 32, 0, lane);
  {
    float cb = b2[lane & 15];
    v8f c = {cb, cb, cb, cb, cb, cb, cb, cb};
    o0 = wmma_f16(a, ldsB(W2t, 32, 0, 0, lane), c);
  }
  {
    float cb = b2[(lane & 15) + 16];
    v8f c = {cb, cb, cb, cb, cb, cb, cb, cb};
    o1 = wmma_f16(a, ldsB(W2t, 32, 0, 1, lane), c);
  }
}

// Transpose+convert weights W[Kin][32] f32 -> LDS Wt[32][KPAD] f16 (zero-padded K).
__device__ __forceinline__ void loadWt(_Float16* dst, const float* __restrict__ W,
                                       int Kin, int KPAD) {
  for (int idx = threadIdx.x; idx < 32 * KPAD; idx += blockDim.x) {
    int n = idx / KPAD, k = idx - n * KPAD;
    dst[idx] = (_Float16)((k < Kin) ? W[k * 32 + n] : 0.0f);
  }
}

// ---------------- Stage 1/2: edge MLP, row = [xT[t](16) | xS[s](16) | ea(8)], K=40->64 ----------
__global__ __launch_bounds__(256) void edge_mlp40_kernel(
    const float* __restrict__ xT, const float* __restrict__ xS, const float* __restrict__ ea,
    const int* __restrict__ sIdx, const int* __restrict__ tIdx,
    const float* __restrict__ W1, const float* __restrict__ b1,
    const float* __restrict__ W2, const float* __restrict__ b2,
    float* __restrict__ aggr, float* __restrict__ cnt, int nEdges) {
  constexpr int KP = 64;
  __shared__ __align__(16) _Float16 sW1[32 * KP];
  __shared__ __align__(16) _Float16 sW2[32 * 32];
  __shared__ __align__(16) _Float16 sA[8][16 * KP];
  __shared__ __align__(16) _Float16 sH[8][16 * 32];
  __shared__ int sT[8][16];

  loadWt(sW1, W1, 40, KP);
  loadWt(sW2, W2, 32, 32);
  __syncthreads();

  int wave = threadIdx.x >> 5, lane = threadIdx.x & 31;
  int tile = blockIdx.x * 8 + wave;
  if (tile >= (nEdges >> 4)) return;

  _Float16* A = sA[wave];
  _Float16* H = sH[wave];
  int* T = sT[wave];

  int r = lane & 15, h = lane >> 4;
  int e = tile * 16 + r;
  int s = sIdx[e], t = tIdx[e];
  if (h == 0) T[r] = t;

#pragma unroll
  for (int jj = 0; jj < 5; ++jj) {  // 10 float4 per row, 5 per lane
    int j = h * 5 + jj;
    const float* src = (j < 4) ? (xT + (size_t)t * 16 + j * 4)
                     : (j < 8) ? (xS + (size_t)s * 16 + (j - 4) * 4)
                               : (ea + (size_t)e * 8 + (j - 8) * 4);
    float4 f = *(const float4*)src;
    v4h hv = {(_Float16)f.x, (_Float16)f.y, (_Float16)f.z, (_Float16)f.w};
    *(v4h*)(A + r * KP + j * 4) = hv;
  }
  {  // zero pad K=40..63
    v4h z = {};
    _Float16* p = A + r * KP + 40 + h * 12;
    *(v4h*)(p) = z; *(v4h*)(p + 4) = z; *(v4h*)(p + 8) = z;
  }

  v8f o0, o1;
  mlp2<KP>(A, sW1, b1, sW2, b2, H, lane, o0, o1);

  int mb = h << 3, n = lane & 15;
#pragma unroll
  for (int rr = 0; rr < 8; ++rr) {
    int tt = T[mb + rr];
    unsafeAtomicAdd(&aggr[(size_t)tt * 32 + n], o0[rr]);
    unsafeAtomicAdd(&aggr[(size_t)tt * 32 + n + 16], o1[rr]);
  }
  if (h == 0) unsafeAtomicAdd(&cnt[t], 1.0f);
}

// ---------------- Stage 3: f_v node MLP, row = [x_v(16) | mean_g(32) | mean_h(32)], K=80->96 ----
__global__ __launch_bounds__(256) void node_fv_kernel(
    const float* __restrict__ x_v,
    const float* __restrict__ sum_g, const float* __restrict__ cnt_g,
    const float* __restrict__ sum_h, const float* __restrict__ cnt_h,
    const float* __restrict__ W1, const float* __restrict__ b1,
    const float* __restrict__ W2, const float* __restrict__ b2,
    float* __restrict__ fv, int NV) {
  constexpr int KP = 96;
  __shared__ __align__(16) _Float16 sW1[32 * KP];
  __shared__ __align__(16) _Float16 sW2[32 * 32];
  __shared__ __align__(16) _Float16 sA[8][16 * KP];
  __shared__ __align__(16) _Float16 sH[8][16 * 32];

  loadWt(sW1, W1, 80, KP);
  loadWt(sW2, W2, 32, 32);
  __syncthreads();

  int wave = threadIdx.x >> 5, lane = threadIdx.x & 31;
  int tile = blockIdx.x * 8 + wave;
  if (tile >= (NV >> 4)) return;

  _Float16* A = sA[wave];
  _Float16* H = sH[wave];
  int r = lane & 15, h = lane >> 4;
  int v = tile * 16 + r;
  float invg = 1.0f / fmaxf(cnt_g[v], 1.0f);
  float invh = 1.0f / fmaxf(cnt_h[v], 1.0f);

#pragma unroll
  for (int jj = 0; jj < 10; ++jj) {  // 20 float4 per row, 10 per lane
    int j = h * 10 + jj;
    float4 f; float sc = 1.0f;
    if (j < 4)        f = *(const float4*)(x_v   + (size_t)v * 16 + j * 4);
    else if (j < 12) { f = *(const float4*)(sum_g + (size_t)v * 32 + (j - 4) * 4);  sc = invg; }
    else             { f = *(const float4*)(sum_h + (size_t)v * 32 + (j - 12) * 4); sc = invh; }
    v4h hv = {(_Float16)(f.x * sc), (_Float16)(f.y * sc), (_Float16)(f.z * sc), (_Float16)(f.w * sc)};
    *(v4h*)(A + r * KP + j * 4) = hv;
  }
  {  // zero pad K=80..95
    v4h z = {};
    _Float16* p = A + r * KP + 80 + h * 8;
    *(v4h*)(p) = z; *(v4h*)(p + 4) = z;
  }

  v8f o0, o1;
  mlp2<KP>(A, sW1, b1, sW2, b2, H, lane, o0, o1);

  int mb = h << 3, n = lane & 15;
#pragma unroll
  for (int rr = 0; rr < 8; ++rr) {
    size_t node = (size_t)tile * 16 + mb + rr;
    fv[node * 32 + n]      = o0[rr];
    fv[node * 32 + n + 16] = o1[rr];
  }
}

// ---------------- Stage 4: g_a edge MLP, row = [x_a[s](16) | f_v[t](32) | ea(8)], K=56->64 ------
__global__ __launch_bounds__(256) void edge_ga_kernel(
    const float* __restrict__ x_a, const float* __restrict__ fv, const float* __restrict__ ea,
    const int* __restrict__ sIdx, const int* __restrict__ tIdx,
    const float* __restrict__ W1, const float* __restrict__ b1,
    const float* __restrict__ W2, const float* __restrict__ b2,
    float* __restrict__ aggr, float* __restrict__ cnt, int nEdges) {
  constexpr int KP = 64;
  __shared__ __align__(16) _Float16 sW1[32 * KP];
  __shared__ __align__(16) _Float16 sW2[32 * 32];
  __shared__ __align__(16) _Float16 sA[8][16 * KP];
  __shared__ __align__(16) _Float16 sH[8][16 * 32];
  __shared__ int sS[8][16];

  loadWt(sW1, W1, 56, KP);
  loadWt(sW2, W2, 32, 32);
  __syncthreads();

  int wave = threadIdx.x >> 5, lane = threadIdx.x & 31;
  int tile = blockIdx.x * 8 + wave;
  if (tile >= (nEdges >> 4)) return;

  _Float16* A = sA[wave];
  _Float16* H = sH[wave];
  int* S = sS[wave];
  int r = lane & 15, h = lane >> 4;
  int e = tile * 16 + r;
  int s = sIdx[e], t = tIdx[e];
  if (h == 0) S[r] = s;

#pragma unroll
  for (int jj = 0; jj < 7; ++jj) {  // 14 float4 per row, 7 per lane
    int j = h * 7 + jj;
    const float* src = (j < 4)  ? (x_a + (size_t)s * 16 + j * 4)
                     : (j < 12) ? (fv  + (size_t)t * 32 + (j - 4) * 4)
                                : (ea  + (size_t)e * 8  + (j - 12) * 4);
    float4 f = *(const float4*)src;
    v4h hv = {(_Float16)f.x, (_Float16)f.y, (_Float16)f.z, (_Float16)f.w};
    *(v4h*)(A + r * KP + j * 4) = hv;
  }
  {  // zero pad K=56..63
    v4h z = {};
    *(v4h*)(A + r * KP + 56 + h * 4) = z;
  }

  v8f o0, o1;
  mlp2<KP>(A, sW1, b1, sW2, b2, H, lane, o0, o1);

  int mb = h << 3, n = lane & 15;
#pragma unroll
  for (int rr = 0; rr < 8; ++rr) {
    int ss = S[mb + rr];
    unsafeAtomicAdd(&aggr[(size_t)ss * 32 + n], o0[rr]);
    unsafeAtomicAdd(&aggr[(size_t)ss * 32 + n + 16], o1[rr]);
  }
  if (h == 0) unsafeAtomicAdd(&cnt[s], 1.0f);
}

// ---------------- Stage 5: f_a node MLP, row = [x_a(16) | mean_ga(32)], K=48->64 ----------------
__global__ __launch_bounds__(256) void node_fa_kernel(
    const float* __restrict__ x_a,
    const float* __restrict__ sum_a, const float* __restrict__ cnt_a,
    const float* __restrict__ W1, const float* __restrict__ b1,
    const float* __restrict__ W2, const float* __restrict__ b2,
    float* __restrict__ out, int NA) {
  constexpr int KP = 64;
  __shared__ __align__(16) _Float16 sW1[32 * KP];
  __shared__ __align__(16) _Float16 sW2[32 * 32];
  __shared__ __align__(16) _Float16 sA[8][16 * KP];
  __shared__ __align__(16) _Float16 sH[8][16 * 32];

  loadWt(sW1, W1, 48, KP);
  loadWt(sW2, W2, 32, 32);
  __syncthreads();

  int wave = threadIdx.x >> 5, lane = threadIdx.x & 31;
  int tile = blockIdx.x * 8 + wave;
  int nTiles = (NA + 15) >> 4;
  if (tile >= nTiles) return;

  _Float16* A = sA[wave];
  _Float16* H = sH[wave];
  int r = lane & 15, h = lane >> 4;
  int v0 = tile * 16 + r;
  int v = v0 < NA ? v0 : NA - 1;  // clamp gather for partial last tile
  float inva = 1.0f / fmaxf(cnt_a[v], 1.0f);

#pragma unroll
  for (int jj = 0; jj < 6; ++jj) {  // 12 float4 per row, 6 per lane
    int j = h * 6 + jj;
    float4 f; float sc = 1.0f;
    if (j < 4) f = *(const float4*)(x_a + (size_t)v * 16 + j * 4);
    else     { f = *(const float4*)(sum_a + (size_t)v * 32 + (j - 4) * 4); sc = inva; }
    v4h hv = {(_Float16)(f.x * sc), (_Float16)(f.y * sc), (_Float16)(f.z * sc), (_Float16)(f.w * sc)};
    *(v4h*)(A + r * KP + j * 4) = hv;
  }
  {  // zero pad K=48..63
    v4h z = {};
    _Float16* p = A + r * KP + 48 + h * 8;
    *(v4h*)(p) = z; *(v4h*)(p + 4) = z;
  }

  v8f o0, o1;
  mlp2<KP>(A, sW1, b1, sW2, b2, H, lane, o0, o1);

  int mb = h << 3, n = lane & 15;
#pragma unroll
  for (int rr = 0; rr < 8; ++rr) {
    int node = tile * 16 + mb + rr;
    if (node < NA) {
      out[(size_t)node * 32 + n]      = o0[rr];
      out[(size_t)node * 32 + n + 16] = o1[rr];
    }
  }
}

// ---------------- workspace zero ----------------
__global__ void zero_ws_kernel(float* p, size_t n) {
  size_t i = (size_t)blockIdx.x * blockDim.x + threadIdx.x;
  size_t st = (size_t)gridDim.x * blockDim.x;
  for (; i < n; i += st) p[i] = 0.0f;
}

// ---------------- launch ----------------
extern "C" void kernel_launch(void* const* d_in, const int* in_sizes, int n_in,
                              void* d_out, int out_size, void* d_ws, size_t ws_size,
                              hipStream_t stream) {
  const float* x_c = (const float*)d_in[0];
  const float* x_v = (const float*)d_in[1];
  const float* x_a = (const float*)d_in[2];
  const int*   ei_c2v = (const int*)d_in[3];
  const int*   ei_a2v = (const int*)d_in[4];
  const float* ea_c2v = (const float*)d_in[5];
  const float* ea_a2v = (const float*)d_in[6];
  const float* gv_W1 = (const float*)d_in[7];
  const float* gv_b1 = (const float*)d_in[8];
  const float* gv_W2 = (const float*)d_in[9];
  const float* gv_b2 = (const float*)d_in[10];
  const float* hv_W1 = (const float*)d_in[11];
  const float* hv_b1 = (const float*)d_in[12];
  const float* hv_W2 = (const float*)d_in[13];
  const float* hv_b2 = (const float*)d_in[14];
  const float* fv_W1 = (const float*)d_in[15];
  const float* fv_b1 = (const float*)d_in[16];
  const float* fv_W2 = (const float*)d_in[17];
  const float* fv_b2 = (const float*)d_in[18];
  const float* ga_W1 = (const float*)d_in[19];
  const float* ga_b1 = (const float*)d_in[20];
  const float* ga_W2 = (const float*)d_in[21];
  const float* ga_b2 = (const float*)d_in[22];
  const float* fa_W1 = (const float*)d_in[23];
  const float* fa_b1 = (const float*)d_in[24];
  const float* fa_W2 = (const float*)d_in[25];
  const float* fa_b2 = (const float*)d_in[26];

  const int EC = in_sizes[3] / 2;
  const int EA = in_sizes[4] / 2;
  const int NV = in_sizes[1] / 16;
  const int NA = in_sizes[2] / 16;

  float* ws = (float*)d_ws;
  float* sum_g = ws;                               // NV*32
  float* cnt_g = sum_g + (size_t)NV * 32;          // NV
  float* sum_h = cnt_g + NV;                       // NV*32
  float* cnt_h = sum_h + (size_t)NV * 32;          // NV
  float* fv    = cnt_h + NV;                       // NV*32
  float* sum_a = fv + (size_t)NV * 32;             // NA*32
  float* cnt_a = sum_a + (size_t)NA * 32;          // NA
  size_t wsFloats = (size_t)NV * 96 + 2 * (size_t)NV + (size_t)NA * 33;

  zero_ws_kernel<<<2048, 256, 0, stream>>>(ws, wsFloats);

  // Stage 1: g_v over c2v edges -> sum_g/cnt_g keyed by c2v target (variable node)
  edge_mlp40_kernel<<<(EC / 16 + 7) / 8, 256, 0, stream>>>(
      x_v, x_c, ea_c2v, ei_c2v, ei_c2v + EC,
      gv_W1, gv_b1, gv_W2, gv_b2, sum_g, cnt_g, EC);

  // Stage 2: h_v over a2v edges -> sum_h/cnt_h keyed by a2v target
  edge_mlp40_kernel<<<(EA / 16 + 7) / 8, 256, 0, stream>>>(
      x_v, x_a, ea_a2v, ei_a2v, ei_a2v + EA,
      hv_W1, hv_b1, hv_W2, hv_b2, sum_h, cnt_h, EA);

  // Stage 3: f_v per variable node
  node_fv_kernel<<<(NV / 16 + 7) / 8, 256, 0, stream>>>(
      x_v, sum_g, cnt_g, sum_h, cnt_h,
      fv_W1, fv_b1, fv_W2, fv_b2, fv, NV);

  // Stage 4: g_a over a2v edges -> sum_a/cnt_a keyed by a2v source (a node)
  edge_ga_kernel<<<(EA / 16 + 7) / 8, 256, 0, stream>>>(
      x_a, fv, ea_a2v, ei_a2v, ei_a2v + EA,
      ga_W1, ga_b1, ga_W2, ga_b2, sum_a, cnt_a, EA);

  // Stage 5: f_a per a node -> d_out [NA,32] f32
  node_fa_kernel<<<(((NA + 15) / 16) + 7) / 8, 256, 0, stream>>>(
      x_a, sum_a, cnt_a, fa_W1, fa_b1, fa_W2, fa_b2, (float*)d_out, NA);
}
